// LocalAttention_17222818857554
// MI455X (gfx1250) — compile-verified, matrix operand order checked by reading, not run on hardware
//
#include <hip/hip_runtime.h>

typedef __bf16 bf16_t;
typedef __attribute__((ext_vector_type(16))) __bf16 v16bf;
typedef __attribute__((ext_vector_type(8)))  __bf16 v8bf;
typedef __attribute__((ext_vector_type(4)))  __bf16 v4bf;
typedef __attribute__((ext_vector_type(8)))  float  v8f;
typedef __attribute__((ext_vector_type(4)))  float  v4f;

constexpr int WSZ  = 256;   // window size
constexpr int EMB  = 128;   // head dim
constexpr int CHK  = 128;   // keys per chunk
constexpr int NW   = 16;    // waves per block
constexpr int PADH = 8;     // LDS row padding in halfs (16B)
constexpr int NTHR = NW * 32;
constexpr int NT   = CHK / 16;                  // 8 key tiles per chunk
constexpr int NIT  = (CHK * EMB) / (4 * NTHR);  // 8 load iters

__device__ __forceinline__ bf16_t f2bf(float f) { return (bf16_t)f; }

__device__ __forceinline__ v16bf cat8(v8bf a, v8bf b) {
  return __builtin_shufflevector(a, b, 0,1,2,3,4,5,6,7,8,9,10,11,12,13,14,15);
}

__global__ __launch_bounds__(NTHR, 1)
void la_kernel(const float* __restrict__ q, const float* __restrict__ k,
               const float* __restrict__ v, float* __restrict__ out, int T) {
  __shared__ __align__(16) bf16_t Kb[CHK][EMB + PADH];          // 128 x 136
  __shared__ __align__(16) bf16_t Vt[EMB][CHK + PADH];          // 128 x 136
  __shared__ __align__(16) bf16_t Pst[NW][16][CHK + PADH];      // 16 x 16 x 136

  const int w    = blockIdx.x;
  const int bh   = blockIdx.y;
  const int tid  = threadIdx.x;
  const int wave = tid >> 5;
  const int lane = tid & 31;
  const int m    = lane & 15;
  const int hi   = lane >> 4;

  const int wstart = w * WSZ;
  const long long baseKV = (long long)bh * T * EMB;

  // scalar (SGPR) wave id for uniform tile-skip branches
  const int wvs = __builtin_amdgcn_readfirstlane(wave);
  const int I0  = wstart + wvs * 16;          // first Q row of this wave (scalar)

  // ---- Q strip A-fragments (registers for whole kernel) ----
  const float* qrow = q + baseKV + (long long)(wstart + wave * 16 + m) * EMB;
  v16bf aq[4];
  #pragma unroll
  for (int kc = 0; kc < 4; ++kc) {
    v8f f0 = *(const v8f*)(qrow + kc * 32 + hi * 8);
    v8f f1 = *(const v8f*)(qrow + kc * 32 + 16 + hi * 8);
    v8bf b0, b1;
    #pragma unroll
    for (int j = 0; j < 8; ++j) { b0[j] = f2bf(f0[j]); b1[j] = f2bf(f1[j]); }
    aq[kc] = cat8(b0, b1);
  }

  v8f oacc[8];
  #pragma unroll
  for (int ft = 0; ft < 8; ++ft) oacc[ft] = {};
  float rmax[8], rsum[8];
  #pragma unroll
  for (int r = 0; r < 8; ++r) { rmax[r] = -1e30f; rsum[r] = 0.0f; }

  const float scale = 0.08838834764831845f;   // 128^-0.5
  const int nchunks = (2 * WSZ) / CHK;        // 4

  for (int ch = 0; ch < nchunks; ++ch) {
    const int jbase = wstart - WSZ + ch * CHK;

    // ---- cooperative load: K chunk -> Kb (bf16), V chunk -> Vt (bf16, transposed)
    #pragma unroll
    for (int it = 0; it < NIT; ++it) {
      int idx  = it * NTHR + tid;
      int key  = idx >> 5;
      int c4   = idx & 31;
      int jabs = jbase + key;
      int jc   = jabs < 0 ? 0 : jabs;        // padded keys masked later
      v4f kf = *(const v4f*)(k + baseKV + (long long)jc * EMB + c4 * 4);
      v4f vf = *(const v4f*)(v + baseKV + (long long)jc * EMB + c4 * 4);
      v4bf kb;
      #pragma unroll
      for (int j = 0; j < 4; ++j) kb[j] = f2bf(kf[j]);
      *(v4bf*)&Kb[key][c4 * 4] = kb;
      #pragma unroll
      for (int j = 0; j < 4; ++j) Vt[c4 * 4 + j][key] = f2bf(vf[j]);
    }
    __syncthreads();

    // ---- register-free prefetch of next chunk (global_prefetch_b8) ----
    if (ch + 1 < nchunks) {
      const int jb2 = jbase + CHK;
      #pragma unroll
      for (int it = 0; it < NIT; ++it) {
        int idx  = it * NTHR + tid;
        int key  = idx >> 5;
        int c4   = idx & 31;
        int jabs = jb2 + key;
        int jc   = jabs < 0 ? 0 : jabs;      // clamp: prefetch may be non-speculative
        const float* kp = k + baseKV + (long long)jc * EMB + c4 * 4;
        const float* vp = v + baseKV + (long long)jc * EMB + c4 * 4;
        __builtin_prefetch(kp, 0, 1);
        __builtin_prefetch(vp, 0, 1);
      }
    }

    // scalar validity flags per 16-col key tile
    bool tv[NT];
    #pragma unroll
    for (int nt = 0; nt < NT; ++nt) {
      int j0 = jbase + nt * 16;
      tv[nt] = (j0 <= I0 + 15) && (j0 + 15 >= I0 - WSZ) && (j0 + 15 >= 0);
    }
    bool anyt = false;
    #pragma unroll
    for (int nt = 0; nt < NT; ++nt) anyt = anyt || tv[nt];

    v8f st[NT];
    if (anyt) {
      float tmax[8];
      #pragma unroll
      for (int r = 0; r < 8; ++r) tmax[r] = -1e30f;

      // ---- S = (Q K^T) * scale, only for tiles with any valid column ----
      #pragma unroll
      for (int nt = 0; nt < NT; ++nt) {
        if (!tv[nt]) continue;
        const int j0 = jbase + nt * 16;
        const bool fullv = (j0 >= 0) && (j0 + 15 <= I0) && (j0 >= I0 + 15 - WSZ);
        v8f acc = {};
        #pragma unroll
        for (int kc = 0; kc < 4; ++kc) {
          v8bf r0 = *(const v8bf*)&Kb[nt * 16 + m][kc * 32 + hi * 8];
          v8bf r1 = *(const v8bf*)&Kb[nt * 16 + m][kc * 32 + 16 + hi * 8];
          v16bf bk = cat8(r0, r1);
          acc = __builtin_amdgcn_wmma_f32_16x16x32_bf16(false, aq[kc], false, bk,
                                                        (short)0, acc, false, false);
        }
        if (fullv) {
          #pragma unroll
          for (int r = 0; r < 8; ++r) {
            float s = acc[r] * scale;
            acc[r] = s;
            tmax[r] = fmaxf(tmax[r], s);
          }
        } else {
          int jcol = j0 + m;
          #pragma unroll
          for (int r = 0; r < 8; ++r) {
            int irow = wstart + wave * 16 + r + 8 * hi;
            float s = acc[r] * scale;
            bool valid = (jcol >= 0) && (irow >= jcol) && (irow - jcol <= WSZ);
            s = valid ? s : -1e30f;
            acc[r] = s;
            tmax[r] = fmaxf(tmax[r], s);
          }
        }
        st[nt] = acc;
      }

      // ---- online softmax update (reduce across the 16-lane half) ----
      #pragma unroll
      for (int r = 0; r < 8; ++r) {
        float mx = tmax[r];
        #pragma unroll
        for (int off = 1; off < 16; off <<= 1) mx = fmaxf(mx, __shfl_xor(mx, off, 32));
        float mnew = fmaxf(rmax[r], mx);
        float corr = __expf(rmax[r] - mnew);
        rmax[r] = mnew;
        float psum = 0.0f;
        #pragma unroll
        for (int nt = 0; nt < NT; ++nt) {
          if (!tv[nt]) continue;
          float p = (st[nt][r] <= -1e29f) ? 0.0f : __expf(st[nt][r] - mnew);
          st[nt][r] = p;
          psum += p;
        }
        #pragma unroll
        for (int off = 1; off < 16; off <<= 1) psum += __shfl_xor(psum, off, 32);
        rsum[r] = rsum[r] * corr + psum;
        #pragma unroll
        for (int ft = 0; ft < 8; ++ft) oacc[ft][r] *= corr;
      }

      // ---- stage P (zero-fill dead tile of a live 32-key group) ----
      #pragma unroll
      for (int nt = 0; nt < NT; ++nt) {
        if (!(tv[nt] || tv[nt ^ 1])) continue;  // group never read by PV
        if (tv[nt]) {
          #pragma unroll
          for (int r = 0; r < 8; ++r)
            Pst[wave][r + 8 * hi][nt * 16 + m] = f2bf(st[nt][r]);
        } else {
          #pragma unroll
          for (int r = 0; r < 8; ++r)
            Pst[wave][r + 8 * hi][nt * 16 + m] = (bf16_t)0.0f;
        }
      }
    }
    __syncthreads();

    // ---- O += P V, skipping fully-dead 32-key groups ----
    if (anyt) {
      #pragma unroll
      for (int kc = 0; kc < CHK / 32; ++kc) {
        if (!(tv[2 * kc] || tv[2 * kc + 1])) continue;
        v8bf p0 = *(const v8bf*)&Pst[wave][m][kc * 32 + hi * 8];
        v8bf p1 = *(const v8bf*)&Pst[wave][m][kc * 32 + 16 + hi * 8];
        v16bf ap = cat8(p0, p1);
        #pragma unroll
        for (int ft = 0; ft < 8; ++ft) {
          v8bf v0 = *(const v8bf*)&Vt[ft * 16 + m][kc * 32 + hi * 8];
          v8bf v1 = *(const v8bf*)&Vt[ft * 16 + m][kc * 32 + 16 + hi * 8];
          v16bf bv = cat8(v0, v1);
          oacc[ft] = __builtin_amdgcn_wmma_f32_16x16x32_bf16(false, ap, false, bv,
                                                             (short)0, oacc[ft], false, false);
        }
      }
    }
    __syncthreads();   // protect Kb/Vt/Pst before next chunk's stores
  }

  // ---- normalize and write out (fp32) ----
  const long long obase = baseKV + (long long)(wstart + wave * 16) * EMB;
  #pragma unroll
  for (int ft = 0; ft < 8; ++ft) {
    #pragma unroll
    for (int r = 0; r < 8; ++r) {
      float val = oacc[ft][r] / rsum[r];
      out[obase + (long long)(r + 8 * hi) * EMB + ft * 16 + m] = val;
    }
  }
}

extern "C" void kernel_launch(void* const* d_in, const int* in_sizes, int n_in,
                              void* d_out, int out_size, void* d_ws, size_t ws_size,
                              hipStream_t stream) {
  const float* q = (const float*)d_in[0];
  const float* k = (const float*)d_in[1];
  const float* v = (const float*)d_in[2];
  float* out = (float*)d_out;
  const int T = 4096;
  const int BH = in_sizes[0] / (T * EMB);      // = 32
  dim3 grid(T / WSZ, BH);                      // (16, 32)
  dim3 block(NTHR);                            // 512 threads = 16 waves
  hipLaunchKernelGGL(la_kernel, grid, block, 0, stream, q, k, v, out, T);
}
